// RnnLayer_28020366639124
// MI455X (gfx1250) — matrix operand order repaired and checked by reading, not compiled, not run on hardware
//
#include <hip/hip_runtime.h>

typedef __attribute__((ext_vector_type(16))) _Float16 v16h;
typedef __attribute__((ext_vector_type(8)))  _Float16 v8h;
typedef __attribute__((ext_vector_type(8)))  float    v8f;
typedef __attribute__((ext_vector_type(4)))  float    v4f;

#define B_   32
#define T_   2048
#define DH_  256
#define NT_  16      // N tiles  = DH/16
#define KF_  8       // K frags  = DH/32
#define LDH  272     // padded LDS row stride in halves (544B: 16B aligned, bank-spread)

// ---------------------------------------------------------------------------
// Pack W_in / W_h (row-major f32 [K=256][N=256]) into WMMA B-fragment order,
// f16. Layout per CDNA5 ISA (16-bit B, 32x16, wave32):
//   fragment f covers K = f*32 .. f*32+31
//   lanes 0-15 : column N = ntile*16 + lane,   K = f*32 + {0..7, 16..23}
//   lanes 16-31: column N = ntile*16 + lane-16, K = f*32 + {8..15, 24..31}
// Packed: wp[w][((ntile*8 + f)*32 + lane)*16 + e]  -> one v16h (32B) per lane.
// ---------------------------------------------------------------------------
__global__ void pack_w_kernel(const float* __restrict__ Wi,
                              const float* __restrict__ Wh,
                              _Float16* __restrict__ wp) {
  int gid   = blockIdx.x * blockDim.x + threadIdx.x;   // 131072 total
  int w     = gid >> 16;                                // 0: W_in, 1: W_h
  int rem   = gid & 0xFFFF;
  int e     = rem & 15;
  int lane  = (rem >> 4) & 31;
  int frag  = (rem >> 9) & 7;
  int ntile = rem >> 12;
  const float* src = w ? Wh : Wi;
  int k   = frag * 32 + ((lane >> 4) << 3) + e + ((e >> 3) << 3);
  int col = (ntile << 4) + (lane & 15);
  wp[(size_t)w * 65536 + (((ntile * KF_ + frag) * 32 + lane) << 4) + e] =
      (_Float16)src[k * DH_ + col];
}

// ---------------------------------------------------------------------------
// Input projection: xp = x @ W_in + b_in, staged into d_out.
// One wave per 16x16 output tile; 8x v_wmma_f32_16x16x32_f16 per wave.
// A fragments built on the fly from f32 x (per ISA 16-bit A 16x32 layout:
//   lane<16: row=lane,   K = f*32 + {0..7, 16..23}
//   lane>=16: row=lane-16, K = f*32 + {8..15, 24..31})
// ---------------------------------------------------------------------------
__global__ void __launch_bounds__(256)
xproj_kernel(const float* __restrict__ x, const float* __restrict__ b_in,
             const _Float16* __restrict__ wp, float* __restrict__ xp) {
  const int lane  = threadIdx.x & 31;
  const int wave  = blockIdx.x * (blockDim.x >> 5) + (threadIdx.x >> 5);
  const int mtile = wave >> 4;            // 0..4095 over B*T rows
  const int ntile = wave & 15;
  const int colL  = lane & 15;
  const int kb    = (lane >> 4) << 3;     // 0 or 8
  const int row   = (mtile << 4) + colL;  // this lane's A row

  v16h bfrag[KF_];
#pragma unroll
  for (int f = 0; f < KF_; ++f)
    bfrag[f] = *(const v16h*)(wp + (((ntile * KF_ + f) * 32 + lane) << 4));

  v8f c = {};
#pragma unroll
  for (int f = 0; f < KF_; ++f) {
    const float* p = x + (size_t)row * DH_ + f * 32 + kb;
    v4f x0 = *(const v4f*)(p);
    v4f x1 = *(const v4f*)(p + 4);
    v4f x2 = *(const v4f*)(p + 16);
    v4f x3 = *(const v4f*)(p + 20);
    v16h a;
#pragma unroll
    for (int i = 0; i < 4; ++i) {
      a[i]      = (_Float16)x0[i];
      a[4 + i]  = (_Float16)x1[i];
      a[8 + i]  = (_Float16)x2[i];
      a[12 + i] = (_Float16)x3[i];
    }
    c = __builtin_amdgcn_wmma_f32_16x16x32_f16(false, a, false, bfrag[f],
                                               (short)0, c, false, false);
  }

  // C layout: VGPR r -> lanes 0-15: (M=r, N=lane); lanes 16-31: (M=8+r, N=lane-16)
  const int col  = (ntile << 4) + colL;
  const float bv = b_in[col];
  const int m0   = (mtile << 4) + ((lane >> 4) << 3);
#pragma unroll
  for (int r = 0; r < 8; ++r)
    xp[(size_t)(m0 + r) * DH_ + col] = c[r] + bv;
}

// ---------------------------------------------------------------------------
// Recurrent scan: 2 workgroups (16 batch rows each) x 16 waves (one N-tile
// each). W_h fragments live in registers for all 2048 steps. h is exchanged
// between waves through a double-buffered LDS tile; one barrier per step.
// xp is read from d_out and overwritten in place with h (same lane -> no race).
// ---------------------------------------------------------------------------
__global__ void __launch_bounds__(512)
rnn_scan_kernel(const float* __restrict__ h0,
                const float* __restrict__ b_h,
                const _Float16* __restrict__ wph,
                float* out) {
  __shared__ __align__(16) _Float16 hbuf[2][16][LDH];

  const int btile = blockIdx.x;            // 0..1 -> batches btile*16..+15
  const int tid   = threadIdx.x;
  const int lane  = tid & 31;
  const int ntile = tid >> 5;              // wave id = N tile
  const int colL  = lane & 15;
  const int kb    = (lane >> 4) << 3;      // 0 or 8
  const int col   = (ntile << 4) + colL;   // H column owned by this lane
  const int mL0   = (lane >> 4) << 3;      // local batch row base: 0 or 8

  v16h bfrag[KF_];                         // W_h, register-resident
#pragma unroll
  for (int f = 0; f < KF_; ++f)
    bfrag[f] = *(const v16h*)(wph + (((ntile * KF_ + f) * 32 + lane) << 4));

  const float bias = b_h[col];

  // seed h_0 (f32 -> f16) into buffer 0
  for (int i = tid; i < 16 * DH_; i += blockDim.x) {
    int r = i >> 8, k = i & 255;
    hbuf[0][r][k] = (_Float16)h0[(size_t)(btile * 16 + r) * DH_ + k];
  }
  __syncthreads();

  const size_t rowstride = (size_t)T_ * DH_;                     // per batch
  float* outp0 = out + (size_t)(btile * 16 + mL0) * rowstride + col;

  for (int t = 0; t < T_; ++t) {
    const int cur = t & 1, nxt = cur ^ 1;

    // A fragments of h_t (16-bit A 16x32 layout), from padded row-major LDS
    v16h a[KF_];
#pragma unroll
    for (int f = 0; f < KF_; ++f) {
      const _Float16* p = &hbuf[cur][colL][f * 32 + kb];
      v8h lo = *(const v8h*)(p);        // K = f*32 + kb + 0..7
      v8h hi = *(const v8h*)(p + 16);   // K = f*32 + kb + 16..23
      a[f] = __builtin_shufflevector(lo, hi,
                                     0, 1, 2, 3, 4, 5, 6, 7,
                                     8, 9, 10, 11, 12, 13, 14, 15);
    }

    v8f c = {};
#pragma unroll
    for (int f = 0; f < KF_; ++f)
      c = __builtin_amdgcn_wmma_f32_16x16x32_f16(false, a[f], false, bfrag[f],
                                                 (short)0, c, false, false);

    // h_new = relu(xp + h@W_h + b); write f32 out, f16 into next LDS buffer
#pragma unroll
    for (int r = 0; r < 8; ++r) {
      float* po = outp0 + (size_t)r * rowstride + (size_t)t * DH_;
      float v = c[r] + *po + bias;      // *po is xp[b,t,col]
      v = v > 0.0f ? v : 0.0f;
      *po = v;                          // overwrite xp with h (same lane)
      hbuf[nxt][mL0 + r][col] = (_Float16)v;
      if (t + 1 < T_) __builtin_prefetch(po + DH_, 0, 3);  // next step's xp
    }
    __syncthreads();
  }
}

extern "C" void kernel_launch(void* const* d_in, const int* in_sizes, int n_in,
                              void* d_out, int out_size, void* d_ws, size_t ws_size,
                              hipStream_t stream) {
  const float* x    = (const float*)d_in[0];
  const float* h0   = (const float*)d_in[1];
  const float* W_in = (const float*)d_in[2];
  const float* b_in = (const float*)d_in[3];
  const float* W_h  = (const float*)d_in[4];
  const float* b_h  = (const float*)d_in[5];
  float*    out = (float*)d_out;
  _Float16* wp  = (_Float16*)d_ws;   // [2][16*8*32*16] packed weights (256 KB)

  pack_w_kernel<<<512, 256, 0, stream>>>(W_in, W_h, wp);
  xproj_kernel<<<8192, 256, 0, stream>>>(x, b_in, wp, out);       // xp -> d_out
  rnn_scan_kernel<<<2, 512, 0, stream>>>(h0, b_h, wp + 65536, out);
}
